// CrossAttentionRefinement_56831007261078
// MI455X (gfx1250) — matrix operand order repaired and checked by reading, not compiled, not run on hardware
//
#include <hip/hip_runtime.h>
#include <math.h>

// ---------------- problem sizes ----------------
#define B_   4
#define N_   1024
#define M_   1024
#define PFD_ 1024
#define AD_  256
#define GFD_ 512
#define H_   4
#define DH_  64
#define NR_  16

#define SCALE_  0.125f               // DH^-0.5
#define COEFF_  (-28.125f)           // -0.5/(CUT/(NR-1))^2
#define DELTA_  (2.0f/15.0f)         // CUT/(NR-1)

typedef __attribute__((ext_vector_type(16))) __bf16 v16bf;
typedef __attribute__((ext_vector_type(8)))  __bf16 v8bf;
typedef __attribute__((ext_vector_type(8)))  float  v8f;

// ---------------- GEMM tiling ----------------
#define TBM 128            // block rows
#define TBN 64             // block cols
#define TBK 32             // K depth per WMMA
#define APAD 8             // pad to 40 bf16 = 80B rows (16B aligned)
#define GT  128            // 4 waves of 32

__device__ __forceinline__ v8f v8f_zero() {
  v8f r;
#pragma unroll
  for (int i = 0; i < 8; ++i) r[i] = 0.f;
  return r;
}

// ---- tile load/store helpers (register-staged double buffering) ----
__device__ __forceinline__ void loadA_g(float4* ar, const float* __restrict__ Ab,
                                        int lda, int rowBase, int k0, int tid) {
#pragma unroll
  for (int i = 0; i < 8; ++i) {              // 128x32 floats
    const int q = tid + i * GT;
    const int r = q >> 3, c = (q & 7) << 2;
    ar[i] = *(const float4*)(Ab + (long)(rowBase + r) * lda + (k0 + c));
  }
}
__device__ __forceinline__ void storeA_s(__bf16 (*As)[TBK + APAD], const float4* ar, int tid) {
#pragma unroll
  for (int i = 0; i < 8; ++i) {
    const int q = tid + i * GT;
    const int r = q >> 3, c = (q & 7) << 2;
    As[r][c + 0] = (__bf16)ar[i].x; As[r][c + 1] = (__bf16)ar[i].y;
    As[r][c + 2] = (__bf16)ar[i].z; As[r][c + 3] = (__bf16)ar[i].w;
  }
}
// NN: B is K x N row-major -> store transposed into Bs[n][k]
__device__ __forceinline__ void loadB_nn_g(float4* br, const float* __restrict__ Bb,
                                           int ldb, int colBase, int k0, int tid) {
#pragma unroll
  for (int i = 0; i < 4; ++i) {              // 32x64 floats
    const int q = tid + i * GT;
    const int r = q >> 4, c = (q & 15) << 2;
    br[i] = *(const float4*)(Bb + (long)(k0 + r) * ldb + (colBase + c));
  }
}
__device__ __forceinline__ void storeB_nn_s(__bf16 (*Bs)[TBK + APAD], const float4* br, int tid) {
#pragma unroll
  for (int i = 0; i < 4; ++i) {
    const int q = tid + i * GT;
    const int r = q >> 4, c = (q & 15) << 2;
    Bs[c + 0][r] = (__bf16)br[i].x; Bs[c + 1][r] = (__bf16)br[i].y;
    Bs[c + 2][r] = (__bf16)br[i].z; Bs[c + 3][r] = (__bf16)br[i].w;
  }
}
// NT: B2 is N x K row-major -> already fragment-major, store direct Bs[n][k]
__device__ __forceinline__ void loadB_nt_g(float4* br, const float* __restrict__ Bb,
                                           int ldb, int colBase, int k0, int tid) {
#pragma unroll
  for (int i = 0; i < 4; ++i) {              // 64x32 floats
    const int q = tid + i * GT;
    const int r = q >> 3, c = (q & 7) << 2;
    br[i] = *(const float4*)(Bb + (long)(colBase + r) * ldb + (k0 + c));
  }
}
__device__ __forceinline__ void storeB_nt_s(__bf16 (*Bs)[TBK + APAD], const float4* br, int tid) {
#pragma unroll
  for (int i = 0; i < 4; ++i) {
    const int q = tid + i * GT;
    const int r = q >> 3, c = (q & 7) << 2;
    Bs[r][c + 0] = (__bf16)br[i].x; Bs[r][c + 1] = (__bf16)br[i].y;
    Bs[r][c + 2] = (__bf16)br[i].z; Bs[r][c + 3] = (__bf16)br[i].w;
  }
}
// A fragment: two contiguous 16B runs per lane (ISA 16-bit A 16x32 layout)
__device__ __forceinline__ v16bf fragA(const __bf16 (*As)[TBK + APAD], int am, int hlf) {
  const v8bf lo = *(const v8bf*)&As[am][hlf * 8];
  const v8bf hi = *(const v8bf*)&As[am][16 + hlf * 8];
  return __builtin_shufflevector(lo, hi, 0, 1, 2, 3, 4, 5, 6, 7,
                                 8, 9, 10, 11, 12, 13, 14, 15);
}
// B fragment: two contiguous 16B runs per lane from Bs[n][k] (same k order as A)
__device__ __forceinline__ v16bf fragB(const __bf16 (*Bs)[TBK + APAD], int n, int hlf) {
  const v8bf lo = *(const v8bf*)&Bs[n][hlf * 8];
  const v8bf hi = *(const v8bf*)&Bs[n][16 + hlf * 8];
  return __builtin_shufflevector(lo, hi, 0, 1, 2, 3, 4, 5, 6, 7,
                                 8, 9, 10, 11, 12, 13, 14, 15);
}

// =====================================================================
// C = A(MxK,row) @ B(KxN,row) [+bias][+resid][gelu]; grid=(N/64, M/128, batch)
// epi: 0 = +bias ; 1 = +bias+resid ; 2 = gelu(+bias)
// =====================================================================
__global__ void __launch_bounds__(GT)
gemm_nn_bf16(const float* __restrict__ A, int lda, long aBatch,
             const float* __restrict__ Bw, int ldb, long bBatch,
             float* __restrict__ C, int ldc, long cBatch,
             const float* __restrict__ bias,
             const float* __restrict__ resid, long rBatch,
             int K, int epi)
{
  __shared__ __attribute__((aligned(16))) __bf16 As[2][TBM][TBK + APAD];
  __shared__ __attribute__((aligned(16))) __bf16 Bs[2][TBN][TBK + APAD];
  const int z = blockIdx.z;
  const float* Ab = A + (long)z * aBatch;
  const float* Bb = Bw + (long)z * bBatch;
  float* Cb = C + (long)z * cBatch;
  const float* Rb = resid ? (resid + (long)z * rBatch) : nullptr;
  const int rowBase = blockIdx.y * TBM;
  const int colBase = blockIdx.x * TBN;
  const int tid = threadIdx.x;
  const int lane = tid & 31, wave = tid >> 5;
  const int l16 = lane & 15, hlf = lane >> 4;

  v8f acc[2][4];
#pragma unroll
  for (int f = 0; f < 2; ++f)
#pragma unroll
    for (int ct = 0; ct < 4; ++ct) acc[f][ct] = v8f_zero();

  const int nT = K / TBK;
  float4 ar[8], br[4];
  loadA_g(ar, Ab, lda, rowBase, 0, tid);
  loadB_nn_g(br, Bb, ldb, colBase, 0, tid);
  storeA_s(As[0], ar, tid);
  storeB_nn_s(Bs[0], br, tid);
  __syncthreads();

  for (int t = 0; t < nT; ++t) {
    const int cur = t & 1;
    const bool more = (t + 1) < nT;
    if (more) {                                // issue next-tile global loads early
      loadA_g(ar, Ab, lda, rowBase, (t + 1) * TBK, tid);
      loadB_nn_g(br, Bb, ldb, colBase, (t + 1) * TBK, tid);
      if (t + 2 < nT)                          // global_prefetch_b8 for t+2
        __builtin_prefetch(Ab + (long)(rowBase + tid) * lda + (t + 2) * TBK, 0, 1);
    }
    v16bf af[2];
#pragma unroll
    for (int f = 0; f < 2; ++f) af[f] = fragA(As[cur], wave * 32 + f * 16 + l16, hlf);
#pragma unroll
    for (int ct = 0; ct < 4; ++ct) {
      const v16bf bv = fragB(Bs[cur], ct * 16 + l16, hlf);
#pragma unroll
      for (int f = 0; f < 2; ++f)
        acc[f][ct] = __builtin_amdgcn_wmma_f32_16x16x32_bf16(
            false, af[f], false, bv, (short)0, acc[f][ct], false, false);
    }
    if (more) {
      storeA_s(As[cur ^ 1], ar, tid);
      storeB_nn_s(Bs[cur ^ 1], br, tid);
    }
    __syncthreads();
  }

#pragma unroll
  for (int f = 0; f < 2; ++f)
#pragma unroll
    for (int ct = 0; ct < 4; ++ct)
#pragma unroll
      for (int r = 0; r < 8; ++r) {
        const int row = rowBase + wave * 32 + f * 16 + r + 8 * hlf;  // C/D layout
        const int col = colBase + ct * 16 + l16;
        float v = acc[f][ct][r];
        if (bias) v += bias[col];
        if (epi == 1 && Rb) v += Rb[(long)row * ldc + col];
        if (epi == 2) v = 0.5f * v * (1.0f + erff(v * 0.70710678118654752f));
        Cb[(long)row * ldc + col] = v;
      }
}

// =====================================================================
// C = A(MxK,row) @ B2(NxK,row)^T ; grid=(Ncols/64, Mrows/128, B)
// epi: 0 = plain ; 1 = *SCALE + geometric RBF bias (attention logits)
//      2 = * 1/clip(temperature)                  (correspondence logits)
// =====================================================================
__global__ void __launch_bounds__(GT)
gemm_nt_bf16(const float* __restrict__ A, int lda, long aBatch,
             const float* __restrict__ B2, int ldb, long bBatch,
             float* __restrict__ C, int ldc, long cBatch,
             int Mrows, int Ncols, int K, int epi, int hIdx,
             const float* __restrict__ src_pts, const float* __restrict__ tgt_pts,
             const float* __restrict__ rbfw, const float* __restrict__ temperature)
{
  __shared__ __attribute__((aligned(16))) __bf16 As[2][TBM][TBK + APAD];
  __shared__ __attribute__((aligned(16))) __bf16 Bs[2][TBN][TBK + APAD];
  const int z = blockIdx.z;
  const float* Ab = A + (long)z * aBatch;
  const float* Bb = B2 + (long)z * bBatch;
  float* Cb = C + (long)z * cBatch;
  const int rowBase = blockIdx.y * TBM;
  const int colBase = blockIdx.x * TBN;
  const int tid = threadIdx.x;
  const int lane = tid & 31, wave = tid >> 5;
  const int l16 = lane & 15, hlf = lane >> 4;

  v8f acc[2][4];
#pragma unroll
  for (int f = 0; f < 2; ++f)
#pragma unroll
    for (int ct = 0; ct < 4; ++ct) acc[f][ct] = v8f_zero();

  const int nT = K / TBK;
  float4 ar[8], br[4];
  loadA_g(ar, Ab, lda, rowBase, 0, tid);
  loadB_nt_g(br, Bb, ldb, colBase, 0, tid);
  storeA_s(As[0], ar, tid);
  storeB_nt_s(Bs[0], br, tid);
  __syncthreads();

  for (int t = 0; t < nT; ++t) {
    const int cur = t & 1;
    const bool more = (t + 1) < nT;
    if (more) {
      loadA_g(ar, Ab, lda, rowBase, (t + 1) * TBK, tid);
      loadB_nt_g(br, Bb, ldb, colBase, (t + 1) * TBK, tid);
      if (t + 2 < nT)
        __builtin_prefetch(Ab + (long)(rowBase + tid) * lda + (t + 2) * TBK, 0, 1);
    }
    v16bf af[2];
#pragma unroll
    for (int f = 0; f < 2; ++f) af[f] = fragA(As[cur], wave * 32 + f * 16 + l16, hlf);
#pragma unroll
    for (int ct = 0; ct < 4; ++ct) {
      const v16bf bv = fragB(Bs[cur], ct * 16 + l16, hlf);
#pragma unroll
      for (int f = 0; f < 2; ++f)
        acc[f][ct] = __builtin_amdgcn_wmma_f32_16x16x32_bf16(
            false, af[f], false, bv, (short)0, acc[f][ct], false, false);
    }
    if (more) {
      storeA_s(As[cur ^ 1], ar, tid);
      storeB_nt_s(Bs[cur ^ 1], br, tid);
    }
    __syncthreads();
  }

  float tscale = 1.0f;
  if (epi == 2) {
    float tv = temperature[0];
    tv = fminf(fmaxf(tv, 0.01f), 1.0f);
    tscale = 1.0f / tv;
  }

#pragma unroll
  for (int f = 0; f < 2; ++f)
#pragma unroll
    for (int ct = 0; ct < 4; ++ct)
#pragma unroll
      for (int r = 0; r < 8; ++r) {
        const int row = rowBase + wave * 32 + f * 16 + r + 8 * hlf;
        const int col = colBase + ct * 16 + l16;
        float v = acc[f][ct][r];
        if (epi == 1) {
          const float* sp = src_pts + ((long)z * Mrows + row) * 3;
          const float* tp = tgt_pts + ((long)z * Ncols + col) * 3;
          const float dx = sp[0] - tp[0], dy = sp[1] - tp[1], dz = sp[2] - tp[2];
          const float dist = sqrtf(fmaxf(dx * dx + dy * dy + dz * dz, 0.f));
          float gb = 0.f;
#pragma unroll
          for (int rr = 0; rr < NR_; ++rr) {
            const float u = dist - (float)rr * DELTA_;
            gb += expf(COEFF_ * u * u) * rbfw[rr * H_ + hIdx];
          }
          v = v * SCALE_ + gb;
        } else {
          v *= tscale;
        }
        Cb[(long)row * ldc + col] = v;
      }
}

// ---------------- block reductions (blockDim = 256) ----------------
__device__ __forceinline__ float blockReduceSum(float v, float* sh) {
  const int t = threadIdx.x;
  sh[t] = v; __syncthreads();
  for (int s = 128; s > 0; s >>= 1) { if (t < s) sh[t] += sh[t + s]; __syncthreads(); }
  const float r = sh[0]; __syncthreads();
  return r;
}
__device__ __forceinline__ float blockReduceMax(float v, float* sh) {
  const int t = threadIdx.x;
  sh[t] = v; __syncthreads();
  for (int s = 128; s > 0; s >>= 1) { if (t < s) sh[t] = fmaxf(sh[t], sh[t + s]); __syncthreads(); }
  const float r = sh[0]; __syncthreads();
  return r;
}

// ---------------- LayerNorm rows (optionally ReLU) ----------------
__global__ void ln_rows_k(const float* __restrict__ in, long inStride,
                          float* __restrict__ out, long outStride,
                          const float* __restrict__ g, const float* __restrict__ bb,
                          int D, int relu)
{
  __shared__ float sh[256];
  const long row = blockIdx.x;
  const float* x = in + row * inStride;
  float* y = out + row * outStride;
  const int t = threadIdx.x;
  float s = 0.f, s2 = 0.f;
  for (int d = t; d < D; d += 256) { const float v = x[d]; s += v; s2 += v * v; }
  s = blockReduceSum(s, sh);
  s2 = blockReduceSum(s2, sh);
  const float mu = s / (float)D;
  const float var = s2 / (float)D - mu * mu;
  const float rstd = rsqrtf(var + 1e-5f);
  for (int d = t; d < D; d += 256) {
    float v = (x[d] - mu) * rstd * g[d] + bb[d];
    if (relu) v = fmaxf(v, 0.f);
    y[d] = v;
  }
}

// ---------------- softmax over last dim (in place) ----------------
__global__ void softmax_rows_k(float* __restrict__ data, int D)
{
  __shared__ float sh[256];
  float* x = data + (long)blockIdx.x * D;
  const int t = threadIdx.x;
  float mx = -3.4e38f;
  for (int d = t; d < D; d += 256) mx = fmaxf(mx, x[d]);
  mx = blockReduceMax(mx, sh);
  float s = 0.f;
  for (int d = t; d < D; d += 256) { const float e = expf(x[d] - mx); x[d] = e; s += e; }
  s = blockReduceSum(s, sh);
  const float inv = 1.0f / s;
  for (int d = t; d < D; d += 256) x[d] *= inv;
}

// ---------------- Sinkhorn: subtract logsumexp over rows / cols ----------------
__global__ void sink_row_k(float* __restrict__ logS, int D)
{
  __shared__ float sh[256];
  float* x = logS + (long)blockIdx.x * D;
  const int t = threadIdx.x;
  float mx = -3.4e38f;
  for (int d = t; d < D; d += 256) mx = fmaxf(mx, x[d]);
  mx = blockReduceMax(mx, sh);
  float s = 0.f;
  for (int d = t; d < D; d += 256) s += expf(x[d] - mx);
  s = blockReduceSum(s, sh);
  const float lse = mx + logf(s);
  for (int d = t; d < D; d += 256) x[d] -= lse;
}

__global__ void sink_col_k(float* __restrict__ logS, int Ncnt, int Mstride)
{
  __shared__ float sh[256];
  const int bm = blockIdx.x;
  const int b = bm / Mstride, m = bm % Mstride;
  float* x = logS + (long)b * Ncnt * Mstride + m;
  const int t = threadIdx.x;
  float mx = -3.4e38f;
  for (int n = t; n < Ncnt; n += 256) mx = fmaxf(mx, x[(long)n * Mstride]);
  mx = blockReduceMax(mx, sh);
  float s = 0.f;
  for (int n = t; n < Ncnt; n += 256) s += expf(x[(long)n * Mstride] - mx);
  s = blockReduceSum(s, sh);
  const float lse = mx + logf(s);
  for (int n = t; n < Ncnt; n += 256) x[(long)n * Mstride] -= lse;
}

// ---------------- S = exp(logS), confidence = rowmax, tcp = S @ tgt_points ----------------
__global__ void s_conf_tcp_k(const float* __restrict__ logS, const float* __restrict__ tgt_pts,
                             float* __restrict__ Sout, float* __restrict__ conf,
                             float* __restrict__ tcp)
{
  __shared__ float sh[256];
  const long row = blockIdx.x;                // b*N + n
  const int b = (int)(row / N_);
  const float* x = logS + row * (long)M_;
  float* so = Sout + row * (long)M_;
  const int t = threadIdx.x;
  float mx = 0.f, a0 = 0.f, a1 = 0.f, a2 = 0.f;
  for (int m = t; m < M_; m += 256) {
    const float sv = expf(x[m]);
    so[m] = sv;
    mx = fmaxf(mx, sv);
    const float* tp = tgt_pts + ((long)b * M_ + m) * 3;
    a0 += sv * tp[0]; a1 += sv * tp[1]; a2 += sv * tp[2];
  }
  mx = blockReduceMax(mx, sh);
  a0 = blockReduceSum(a0, sh);
  a1 = blockReduceSum(a1, sh);
  a2 = blockReduceSum(a2, sh);
  if (t == 0) {
    conf[row] = mx;
    tcp[row * 3 + 0] = a0; tcp[row * 3 + 1] = a1; tcp[row * 3 + 2] = a2;
  }
}

// ---------------- Kabsch reductions: wsum, centroids, covariance ----------------
__global__ void kabsch_reduce_k(const float* __restrict__ conf,
                                const float* __restrict__ src_pts,
                                const float* __restrict__ tcp,
                                float* __restrict__ kab)
{
  __shared__ float sh[256];
  const int b = blockIdx.x, t = threadIdx.x;
  const float* cf = conf + (long)b * N_;
  const float* sp = src_pts + (long)b * N_ * 3;
  const float* tp = tcp + (long)b * N_ * 3;
  float s = 0.f;
  for (int n = t; n < N_; n += 256) s += cf[n];
  const float wsum = blockReduceSum(s, sh);
  const float winv = 1.0f / (wsum + 1e-8f);
  float cs[3], ct[3];
  for (int c = 0; c < 3; ++c) {
    float a = 0.f, d = 0.f;
    for (int n = t; n < N_; n += 256) {
      const float w = cf[n] * winv;
      a += w * sp[n * 3 + c];
      d += w * tp[n * 3 + c];
    }
    cs[c] = blockReduceSum(a, sh);
    ct[c] = blockReduceSum(d, sh);
  }
  for (int c = 0; c < 3; ++c)
    for (int d = 0; d < 3; ++d) {
      float a = 0.f;
      for (int n = t; n < N_; n += 256) {
        const float w = cf[n] * winv;
        a += w * (sp[n * 3 + c] - cs[c]) * (tp[n * 3 + d] - ct[d]);
      }
      const float h = blockReduceSum(a, sh);
      if (t == 0) kab[b * 16 + 6 + c * 3 + d] = h;
    }
  if (t == 0) {
    for (int c = 0; c < 3; ++c) { kab[b * 16 + c] = cs[c]; kab[b * 16 + 3 + c] = ct[c]; }
    kab[b * 16 + 15] = wsum;
  }
}

// ---------------- 3x3 SVD (Jacobi on H^T H) + Kabsch R,t ----------------
__device__ __forceinline__ float det3f(const float m[3][3]) {
  return m[0][0] * (m[1][1] * m[2][2] - m[1][2] * m[2][1])
       - m[0][1] * (m[1][0] * m[2][2] - m[1][2] * m[2][0])
       + m[0][2] * (m[1][0] * m[2][1] - m[1][1] * m[2][0]);
}

__global__ void kabsch_svd_k(const float* __restrict__ kab,
                             float* __restrict__ Rout, float* __restrict__ tout)
{
  const int b = threadIdx.x;
  if (b >= B_) return;
  const float* kk = kab + b * 16;
  const float cs[3] = {kk[0], kk[1], kk[2]};
  const float ct[3] = {kk[3], kk[4], kk[5]};
  float Hm[3][3];
  for (int i = 0; i < 3; ++i)
    for (int j = 0; j < 3; ++j) Hm[i][j] = kk[6 + i * 3 + j];

  // K = H^T H, Jacobi eigen-decomposition -> V (right singular vectors)
  float Ks[3][3], V[3][3] = {{1, 0, 0}, {0, 1, 0}, {0, 0, 1}};
  for (int i = 0; i < 3; ++i)
    for (int j = 0; j < 3; ++j) {
      float a = 0.f;
      for (int c = 0; c < 3; ++c) a += Hm[c][i] * Hm[c][j];
      Ks[i][j] = a;
    }
  const int ps[3] = {0, 0, 1}, qs[3] = {1, 2, 2};
  for (int sweep = 0; sweep < 30; ++sweep) {
    const int p = ps[sweep % 3], q = qs[sweep % 3];
    const float apq = Ks[p][q];
    if (fabsf(apq) > 1e-14f) {
      const float tau = (Ks[q][q] - Ks[p][p]) / (2.f * apq);
      const float tt = (tau >= 0.f ? 1.f : -1.f) / (fabsf(tau) + sqrtf(1.f + tau * tau));
      const float c = rsqrtf(1.f + tt * tt);
      const float s = tt * c;
      for (int i = 0; i < 3; ++i) { const float kp = Ks[i][p], kq = Ks[i][q]; Ks[i][p] = c * kp - s * kq; Ks[i][q] = s * kp + c * kq; }
      for (int i = 0; i < 3; ++i) { const float kp = Ks[p][i], kq = Ks[q][i]; Ks[p][i] = c * kp - s * kq; Ks[q][i] = s * kp + c * kq; }
      for (int i = 0; i < 3; ++i) { const float vp = V[i][p], vq = V[i][q]; V[i][p] = c * vp - s * vq; V[i][q] = s * vp + c * vq; }
    }
  }
  float lam[3] = {Ks[0][0], Ks[1][1], Ks[2][2]};
  int idx[3] = {0, 1, 2};
  for (int a = 0; a < 2; ++a)
    for (int c = 0; c < 2 - a; ++c)
      if (lam[idx[c]] < lam[idx[c + 1]]) { const int tmp = idx[c]; idx[c] = idx[c + 1]; idx[c + 1] = tmp; }
  float Vs[3][3];
  for (int i = 0; i < 3; ++i)
    for (int j = 0; j < 3; ++j) Vs[i][j] = V[i][idx[j]];

  // U columns: u_j = H v_j, orthonormalized
  float U[3][3];
  for (int j = 0; j < 3; ++j)
    for (int i = 0; i < 3; ++i) {
      float a = 0.f;
      for (int k2 = 0; k2 < 3; ++k2) a += Hm[i][k2] * Vs[k2][j];
      U[i][j] = a;
    }
  float n0 = sqrtf(U[0][0] * U[0][0] + U[1][0] * U[1][0] + U[2][0] * U[2][0]);
  if (n0 > 1e-12f) { for (int i = 0; i < 3; ++i) U[i][0] /= n0; }
  else { U[0][0] = 1.f; U[1][0] = 0.f; U[2][0] = 0.f; }
  float d01 = U[0][1] * U[0][0] + U[1][1] * U[1][0] + U[2][1] * U[2][0];
  for (int i = 0; i < 3; ++i) U[i][1] -= d01 * U[i][0];
  float n1 = sqrtf(U[0][1] * U[0][1] + U[1][1] * U[1][1] + U[2][1] * U[2][1]);
  if (n1 > 1e-12f) { for (int i = 0; i < 3; ++i) U[i][1] /= n1; }
  else {
    float ax = fabsf(U[0][0]), ay = fabsf(U[1][0]), az = fabsf(U[2][0]);
    float e[3] = {0.f, 0.f, 0.f};
    if (ax <= ay && ax <= az) e[0] = 1.f; else if (ay <= az) e[1] = 1.f; else e[2] = 1.f;
    float c0 = e[1] * U[2][0] - e[2] * U[1][0];
    float c1 = e[2] * U[0][0] - e[0] * U[2][0];
    float c2 = e[0] * U[1][0] - e[1] * U[0][0];
    float nn = sqrtf(c0 * c0 + c1 * c1 + c2 * c2);
    U[0][1] = c0 / nn; U[1][1] = c1 / nn; U[2][1] = c2 / nn;
  }
  float cx = U[1][0] * U[2][1] - U[2][0] * U[1][1];
  float cy = U[2][0] * U[0][1] - U[0][0] * U[2][1];
  float cz = U[0][0] * U[1][1] - U[1][0] * U[0][1];
  float d02 = U[0][2] * U[0][0] + U[1][2] * U[1][0] + U[2][2] * U[2][0];
  float d12 = U[0][2] * U[0][1] + U[1][2] * U[1][1] + U[2][2] * U[2][1];
  for (int i = 0; i < 3; ++i) U[i][2] -= d02 * U[i][0] + d12 * U[i][1];
  float n2 = sqrtf(U[0][2] * U[0][2] + U[1][2] * U[1][2] + U[2][2] * U[2][2]);
  if (n2 > 1e-8f) { for (int i = 0; i < 3; ++i) U[i][2] /= n2; }
  else { U[0][2] = cx; U[1][2] = cy; U[2][2] = cz; }

  const float d = det3f(Vs) * det3f(U);       // det(V U^T)
  float R[3][3];
  for (int i = 0; i < 3; ++i)
    for (int k2 = 0; k2 < 3; ++k2)
      R[i][k2] = Vs[i][0] * U[k2][0] + Vs[i][1] * U[k2][1] + d * Vs[i][2] * U[k2][2];
  for (int i = 0; i < 3; ++i) {
    for (int k2 = 0; k2 < 3; ++k2) Rout[b * 9 + i * 3 + k2] = R[i][k2];
    tout[b * 3 + i] = ct[i] - (R[i][0] * cs[0] + R[i][1] * cs[1] + R[i][2] * cs[2]);
  }
}

// ---------------- global-feature head helpers ----------------
__global__ void gmax_k(const float* __restrict__ x, float* __restrict__ gmax)
{
  const int idx = blockIdx.x * blockDim.x + threadIdx.x;
  if (idx >= B_ * AD_) return;
  const int b = idx / AD_, d = idx % AD_;
  const float* xp = x + (long)b * N_ * AD_ + d;
  float m = -3.4e38f;
  for (int n = 0; n < N_; ++n) m = fmaxf(m, xp[(long)n * AD_]);
  gmax[idx] = m;
}

__global__ void gproj_k(const float* __restrict__ gmax, const float* __restrict__ gp_w,
                        const float* __restrict__ gp_b, float* __restrict__ gproj)
{
  const int idx = blockIdx.x * blockDim.x + threadIdx.x;
  if (idx >= B_ * GFD_) return;
  const int b = idx / GFD_, g2 = idx % GFD_;
  const float* xr = gmax + b * AD_;
  float a = gp_b[g2];
  for (int k2 = 0; k2 < AD_; ++k2) a += xr[k2] * gp_w[(long)k2 * GFD_ + g2];
  gproj[idx] = a;
}

__global__ void copy_tgt_k(const float* __restrict__ tgt_global, float* __restrict__ comb)
{
  const int idx = blockIdx.x * blockDim.x + threadIdx.x;
  if (idx >= B_ * GFD_) return;
  const int b = idx / GFD_, g2 = idx % GFD_;
  comb[(long)b * (2 * GFD_) + GFD_ + g2] = tgt_global[idx];
}

// =====================================================================
// Launcher
// =====================================================================
extern "C" void kernel_launch(void* const* d_in, const int* in_sizes, int n_in,
                              void* d_out, int out_size, void* d_ws, size_t ws_size,
                              hipStream_t stream)
{
  (void)in_sizes; (void)n_in; (void)out_size; (void)ws_size;
  const float* src_feats  = (const float*)d_in[0];
  const float* tgt_feats  = (const float*)d_in[1];
  const float* src_points = (const float*)d_in[2];
  const float* tgt_points = (const float*)d_in[3];
  const float* tgt_global = (const float*)d_in[4];
  const float* sfp_w = (const float*)d_in[5];  const float* sfp_b  = (const float*)d_in[6];
  const float* sfp_g = (const float*)d_in[7];  const float* sfp_bb = (const float*)d_in[8];
  const float* tfp_w = (const float*)d_in[9];  const float* tfp_b  = (const float*)d_in[10];
  const float* tfp_g = (const float*)d_in[11]; const float* tfp_bb = (const float*)d_in[12];
  const float* rbf_w = (const float*)d_in[13];
  const float* qw = (const float*)d_in[14]; const float* qb = (const float*)d_in[15];
  const float* kw = (const float*)d_in[16]; const float* kb = (const float*)d_in[17];
  const float* vw = (const float*)d_in[18]; const float* vb = (const float*)d_in[19];
  const float* ow = (const float*)d_in[20]; const float* ob = (const float*)d_in[21];
  const float* n1_g = (const float*)d_in[22]; const float* n1_b = (const float*)d_in[23];
  const float* f1_w = (const float*)d_in[24]; const float* f1_b = (const float*)d_in[25];
  const float* f2_w = (const float*)d_in[26]; const float* f2_b = (const float*)d_in[27];
  const float* n2_g = (const float*)d_in[28]; const float* n2_b = (const float*)d_in[29];
  const float* scp_w = (const float*)d_in[30]; const float* scp_b = (const float*)d_in[31];
  const float* tcp_w = (const float*)d_in[32]; const float* tcp_b = (const float*)d_in[33];
  const float* temperature = (const float*)d_in[34];
  const float* gp_w = (const float*)d_in[35]; const float* gp_b  = (const float*)d_in[36];
  const float* gp_g = (const float*)d_in[37]; const float* gp_bb = (const float*)d_in[38];

  // workspace layout (floats)
  float* ws = (float*)d_ws;
  const long SZ = (long)B_ * N_ * AD_;          // 1,048,576
  float* sproj  = ws;
  float* tproj  = sproj + SZ;
  float* xq     = tproj + SZ;                   // later reused as attn buffer
  float* xc     = xq + SZ;                      // later reused as FFN-LN buffer
  float* qbuf   = xc + SZ;
  float* kbuf   = qbuf + SZ;
  float* vbuf   = kbuf + SZ;
  float* xbuf   = vbuf + SZ;
  float* ffn1   = xbuf + SZ;                    // B*N*4AD
  float* scorr  = ffn1 + (long)B_ * N_ * 4 * AD_;
  float* tcorr  = scorr + SZ;
  float* logits = tcorr + SZ;                   // B*H*N*M
  float* logS   = logits + (long)B_ * H_ * N_ * M_;  // B*N*M
  float* tcp    = logS + (long)B_ * N_ * M_;    // B*N*3
  float* kab    = tcp + (long)B_ * N_ * 3;      // B*16
  float* gmaxb  = kab + B_ * 16;                // B*AD
  float* gprojb = gmaxb + B_ * AD_;             // B*GFD
  float* attn   = xq;
  float* hbuf   = xc;

  // outputs
  float* out = (float*)d_out;
  float* outR    = out;                         // B*9
  float* outT    = out + 36;                    // B*3
  float* outComb = out + 48;                    // B*2*GFD
  float* outConf = out + 48 + (long)B_ * 2 * GFD_;   // B*N
  float* outS    = outConf + (long)B_ * N_;     // B*N*M

  const dim3 gblk(GT);

  // 1) feature projections + LN + ReLU
  gemm_nn_bf16<<<dim3(AD_ / TBN, (B_ * N_) / TBM, 1), gblk, 0, stream>>>(
      src_feats, PFD_, 0, sfp_w, AD_, 0, sproj, AD_, 0, sfp_b, nullptr, 0, PFD_, 0);
  gemm_nn_bf16<<<dim3(AD_ / TBN, (B_ * N_) / TBM, 1), gblk, 0, stream>>>(
      tgt_feats, PFD_, 0, tfp_w, AD_, 0, tproj, AD_, 0, tfp_b, nullptr, 0, PFD_, 0);
  ln_rows_k<<<B_ * N_, 256, 0, stream>>>(sproj, AD_, sproj, AD_, sfp_g, sfp_bb, AD_, 1);
  ln_rows_k<<<B_ * M_, 256, 0, stream>>>(tproj, AD_, tproj, AD_, tfp_g, tfp_bb, AD_, 1);

  // 2) pre-attention LN (shared params)
  ln_rows_k<<<B_ * N_, 256, 0, stream>>>(sproj, AD_, xq, AD_, n1_g, n1_b, AD_, 0);
  ln_rows_k<<<B_ * M_, 256, 0, stream>>>(tproj, AD_, xc, AD_, n1_g, n1_b, AD_, 0);

  // 3) Q/K/V projections
  gemm_nn_bf16<<<dim3(AD_ / TBN, (B_ * N_) / TBM, 1), gblk, 0, stream>>>(
      xq, AD_, 0, qw, AD_, 0, qbuf, AD_, 0, qb, nullptr, 0, AD_, 0);
  gemm_nn_bf16<<<dim3(AD_ / TBN, (B_ * M_) / TBM, 1), gblk, 0, stream>>>(
      xc, AD_, 0, kw, AD_, 0, kbuf, AD_, 0, kb, nullptr, 0, AD_, 0);
  gemm_nn_bf16<<<dim3(AD_ / TBN, (B_ * M_) / TBM, 1), gblk, 0, stream>>>(
      xc, AD_, 0, vw, AD_, 0, vbuf, AD_, 0, vb, nullptr, 0, AD_, 0);

  // 4) logits = q k^T * SCALE + geometric RBF bias (fused epilogue)
  for (int h = 0; h < H_; ++h) {
    gemm_nt_bf16<<<dim3(M_ / TBN, N_ / TBM, B_), gblk, 0, stream>>>(
        qbuf + h * DH_, AD_, (long)N_ * AD_,
        kbuf + h * DH_, AD_, (long)M_ * AD_,
        logits + (long)h * N_ * M_, M_, (long)H_ * N_ * M_,
        N_, M_, DH_, 1, h, src_points, tgt_points, rbf_w, nullptr);
  }

  // 5) softmax over m
  softmax_rows_k<<<B_ * H_ * N_, 256, 0, stream>>>(logits, M_);

  // 6) attn = p @ v   (into attn = xq)
  for (int h = 0; h < H_; ++h) {
    gemm_nn_bf16<<<dim3(DH_ / TBN, N_ / TBM, B_), gblk, 0, stream>>>(
        logits + (long)h * N_ * M_, M_, (long)H_ * N_ * M_,
        vbuf + h * DH_, AD_, (long)M_ * AD_,
        attn + h * DH_, AD_, (long)N_ * AD_,
        nullptr, nullptr, 0, M_, 0);
  }

  // 7) x = src_proj + attn @ ow + ob
  gemm_nn_bf16<<<dim3(AD_ / TBN, (B_ * N_) / TBM, 1), gblk, 0, stream>>>(
      attn, AD_, 0, ow, AD_, 0, xbuf, AD_, 0, ob, sproj, 0, AD_, 1);

  // 8) FFN: x = x + gelu(LN(x) @ f1 + b1) @ f2 + b2
  ln_rows_k<<<B_ * N_, 256, 0, stream>>>(xbuf, AD_, hbuf, AD_, n2_g, n2_b, AD_, 0);
  gemm_nn_bf16<<<dim3((4 * AD_) / TBN, (B_ * N_) / TBM, 1), gblk, 0, stream>>>(
      hbuf, AD_, 0, f1_w, 4 * AD_, 0, ffn1, 4 * AD_, 0, f1_b, nullptr, 0, AD_, 2);
  gemm_nn_bf16<<<dim3(AD_ / TBN, (B_ * N_) / TBM, 1), gblk, 0, stream>>>(
      ffn1, 4 * AD_, 0, f2_w, AD_, 0, xbuf, AD_, 0, f2_b, xbuf, 0, 4 * AD_, 1);

  // 9) correspondence projections + logS (with 1/tau fused)
  gemm_nn_bf16<<<dim3(AD_ / TBN, (B_ * N_) / TBM, 1), gblk, 0, stream>>>(
      xbuf, AD_, 0, scp_w, AD_, 0, scorr, AD_, 0, scp_b, nullptr, 0, AD_, 0);
  gemm_nn_bf16<<<dim3(AD_ / TBN, (B_ * M_) / TBM, 1), gblk, 0, stream>>>(
      tproj, AD_, 0, tcp_w, AD_, 0, tcorr, AD_, 0, tcp_b, nullptr, 0, AD_, 0);
  gemm_nt_bf16<<<dim3(M_ / TBN, N_ / TBM, B_), gblk, 0, stream>>>(
      scorr, AD_, (long)N_ * AD_, tcorr, AD_, (long)M_ * AD_,
      logS, M_, (long)N_ * M_,
      N_, M_, AD_, 2, 0, nullptr, nullptr, nullptr, temperature);

  // 10) 3-step log-domain Sinkhorn
  for (int it = 0; it < 3; ++it) {
    sink_row_k<<<B_ * N_, 256, 0, stream>>>(logS, M_);
    sink_col_k<<<B_ * M_, 256, 0, stream>>>(logS, N_, M_);
  }

  // 11) S, confidence, corresponding points
  s_conf_tcp_k<<<B_ * N_, 256, 0, stream>>>(logS, tgt_points, outS, outConf, tcp);

  // 12) weighted Kabsch
  kabsch_reduce_k<<<B_, 256, 0, stream>>>(outConf, src_points, tcp, kab);
  kabsch_svd_k<<<1, 32, 0, stream>>>(kab, outR, outT);

  // 13) global feature head + concat
  gmax_k<<<(B_ * AD_ + 255) / 256, 256, 0, stream>>>(xbuf, gmaxb);
  gproj_k<<<(B_ * GFD_ + 255) / 256, 256, 0, stream>>>(gmaxb, gp_w, gp_b, gprojb);
  ln_rows_k<<<B_, 256, 0, stream>>>(gprojb, GFD_, outComb, 2 * GFD_, gp_g, gp_bb, GFD_, 1);
  copy_tgt_k<<<(B_ * GFD_ + 255) / 256, 256, 0, stream>>>(tgt_global, outComb);
}
